// RNN_Embedding_82016695484446
// MI455X (gfx1250) — matrix operand order repaired and checked by reading, not compiled, not run on hardware
//
#include <hip/hip_runtime.h>

// ---------------------------------------------------------------------------
// CDNA5 (gfx1250) implementation of the attention-GRU scan.
//   - bf16 WMMA (v_wmma_f32_16x16x32_bf16) for all GEMMs, f32 accumulate
//   - x converted once to bf16 (134 MB) so the per-step einsum streams from
//     the 192 MB L2 instead of HBM (einsum = dominant cost: 512 x 134 MB)
//   - deterministic: no FP atomics, everything rewritten every call
// ---------------------------------------------------------------------------

typedef __attribute__((ext_vector_type(16))) __bf16 v16bf;
typedef __attribute__((ext_vector_type(8)))  __bf16 bf16x8;
typedef __attribute__((ext_vector_type(4)))  __bf16 bf16x4;
typedef __attribute__((ext_vector_type(8)))  float  v8f;
typedef __attribute__((ext_vector_type(4)))  float  f32x4;
typedef __attribute__((ext_vector_type(4)))  unsigned int u32x4;

#define NB   256   // batch
#define NH   512   // hidden = inputs = T
#define NT   512

static __device__ __forceinline__ __bf16 f2bf(float f) {
  unsigned u; __builtin_memcpy(&u, &f, 4);
  u += 0x7FFFu + ((u >> 16) & 1u);                 // round-to-nearest-even
  unsigned short s = (unsigned short)(u >> 16);
  __bf16 b; __builtin_memcpy(&b, &s, 2); return b;
}
static __device__ __forceinline__ float lo2f(unsigned u) {
  unsigned v = u << 16; float f; __builtin_memcpy(&f, &v, 4); return f;
}
static __device__ __forceinline__ float hi2f(unsigned u) {
  unsigned v = u & 0xFFFF0000u; float f; __builtin_memcpy(&f, &v, 4); return f;
}
static __device__ __forceinline__ float sigf(float x) {
  return 1.0f / (1.0f + __expf(-x));
}

// -------------------------------- f32 -> bf16 (4 elems / thread) -----------
__global__ __launch_bounds__(256) void cvt4_f32_bf16(const float* __restrict__ src,
                                                     __bf16* __restrict__ dst, int n4) {
  int i = blockIdx.x * 256 + threadIdx.x;
  if (i >= n4) return;
  f32x4 v = *(const f32x4*)(src + 4 * (size_t)i);
  bf16x4 o;
  o[0] = f2bf(v[0]); o[1] = f2bf(v[1]); o[2] = f2bf(v[2]); o[3] = f2bf(v[3]);
  *(bf16x4*)(dst + 4 * (size_t)i) = o;
}

// -------------------------------- h = 0, axh[:,512:] = 0 -------------------
__global__ __launch_bounds__(256) void init_state(float* __restrict__ h,
                                                  __bf16* __restrict__ axh) {
  int idx = blockIdx.x * 256 + threadIdx.x;      // 131072
  h[idx] = 0.0f;
  int b = idx >> 9, j = idx & 511;
  axh[((size_t)b << 10) + 512 + j] = f2bf(0.0f);
}

// ----------- gather x_t column into axh[:, :512] and acomb[:, :512] --------
__global__ __launch_bounds__(256) void prep_xt(const __bf16* __restrict__ xbf,
                                               __bf16* __restrict__ axh,
                                               __bf16* __restrict__ acomb, int t) {
  int b = blockIdx.x;
#pragma unroll
  for (int rep = 0; rep < 2; ++rep) {
    int i = threadIdx.x + rep * 256;
    __bf16 v = xbf[((size_t)b << 18) + ((size_t)i << 9) + t];
    axh[((size_t)b << 10) + i]   = v;
    acomb[((size_t)b << 10) + i] = v;
  }
}

// ------------------------- generic bf16 WMMA GEMM --------------------------
// C[M,N] = A[M,K](bf16, row-major, lda) x W[N,K](bf16, K-contiguous).T + bias
// one wave computes a 32x32 tile; mode bit0 = relu, bit1 = bf16 output
__global__ __launch_bounds__(256) void gemm_wmma(
    const __bf16* __restrict__ A, int lda,
    const __bf16* __restrict__ W,
    const float* __restrict__ bias,
    void* __restrict__ Cout, int ldc,
    int M, int N, int K, int mode) {
  const int wave = threadIdx.x >> 5;
  const int lane = threadIdx.x & 31;
  const int tiles_n = N >> 5;
  const int tile = blockIdx.x * 8 + wave;
  const int mt = (tile / tiles_n) << 5;
  const int nt = (tile % tiles_n) << 5;
  if (mt >= M) return;
  const int lm   = lane & 15;
  const int hk8  = (lane >> 4) << 4;   // A frag: lanes 0-15 K 0..15; 16-31 K 16..31
  const int hk16 = (lane >> 4) << 4;   // B frag: lanes 0-15 K 0..15; 16-31 K 16..31
  const __bf16* a0 = A + (size_t)(mt + lm) * lda + hk8;
  const __bf16* a1 = a0 + (size_t)16 * lda;
  const __bf16* b0 = W + (size_t)(nt + lm) * K + hk16;
  const __bf16* b1 = b0 + (size_t)16 * K;
  v8f acc00 = {}, acc01 = {}, acc10 = {}, acc11 = {};
  for (int k = 0; k < K; k += 32) {
    bf16x8 a0l = *(const bf16x8*)(a0 + k);
    bf16x8 a0h = *(const bf16x8*)(a0 + k + 8);
    bf16x8 a1l = *(const bf16x8*)(a1 + k);
    bf16x8 a1h = *(const bf16x8*)(a1 + k + 8);
    v16bf fa0 = __builtin_shufflevector(a0l, a0h, 0,1,2,3,4,5,6,7,8,9,10,11,12,13,14,15);
    v16bf fa1 = __builtin_shufflevector(a1l, a1h, 0,1,2,3,4,5,6,7,8,9,10,11,12,13,14,15);
    v16bf fb0 = *(const v16bf*)(b0 + k);
    v16bf fb1 = *(const v16bf*)(b1 + k);
    acc00 = __builtin_amdgcn_wmma_f32_16x16x32_bf16(false, fa0, false, fb0, (short)0, acc00, false, false);
    acc01 = __builtin_amdgcn_wmma_f32_16x16x32_bf16(false, fa0, false, fb1, (short)0, acc01, false, false);
    acc10 = __builtin_amdgcn_wmma_f32_16x16x32_bf16(false, fa1, false, fb0, (short)0, acc10, false, false);
    acc11 = __builtin_amdgcn_wmma_f32_16x16x32_bf16(false, fa1, false, fb1, (short)0, acc11, false, false);
  }
  const int n0 = nt + lm, n1 = n0 + 16;
  const int mrow = mt + ((lane >> 4) << 3);
  const float bi0 = bias[n0], bi1 = bias[n1];
#pragma unroll
  for (int r = 0; r < 8; ++r) {
    float v00 = acc00[r] + bi0, v01 = acc01[r] + bi1;
    float v10 = acc10[r] + bi0, v11 = acc11[r] + bi1;
    if (mode & 1) {
      v00 = fmaxf(v00, 0.f); v01 = fmaxf(v01, 0.f);
      v10 = fmaxf(v10, 0.f); v11 = fmaxf(v11, 0.f);
    }
    const int m0 = mrow + r, m1 = m0 + 16;
    if (mode & 2) {
      __bf16* C = (__bf16*)Cout;
      C[(size_t)m0 * ldc + n0] = f2bf(v00); C[(size_t)m0 * ldc + n1] = f2bf(v01);
      C[(size_t)m1 * ldc + n0] = f2bf(v10); C[(size_t)m1 * ldc + n1] = f2bf(v11);
    } else {
      float* C = (float*)Cout;
      C[(size_t)m0 * ldc + n0] = v00; C[(size_t)m0 * ldc + n1] = v01;
      C[(size_t)m1 * ldc + n0] = v10; C[(size_t)m1 * ldc + n1] = v11;
    }
  }
}

// --------------------------- row softmax (512 wide), in place --------------
__global__ __launch_bounds__(256) void softmax512(float* __restrict__ s) {
  __shared__ float red[256];
  float* row = s + ((size_t)blockIdx.x << 9);
  int tid = threadIdx.x;
  float a = row[tid], c = row[tid + 256];
  red[tid] = fmaxf(a, c);
  __syncthreads();
  for (int off = 128; off > 0; off >>= 1) {
    if (tid < off) red[tid] = fmaxf(red[tid], red[tid + off]);
    __syncthreads();
  }
  float mx = red[0];
  __syncthreads();
  float e0 = __expf(a - mx), e1 = __expf(c - mx);
  red[tid] = e0 + e1;
  __syncthreads();
  for (int off = 128; off > 0; off >>= 1) {
    if (tid < off) red[tid] += red[tid + off];
    __syncthreads();
  }
  float inv = 1.0f / red[0];
  row[tid] = e0 * inv;
  row[tid + 256] = e1 * inv;
}

// --------- ctx partial: ctxp[half][b][t'] = sum_{m in half} aw*xbf ---------
// block = (b, half) of 64 threads; thread handles 8 contiguous t' (16B loads)
__global__ __launch_bounds__(64) void einsum_ctx(const float* __restrict__ aw,
                                                 const __bf16* __restrict__ xbf,
                                                 float* __restrict__ ctxp) {
  const int b = blockIdx.x, half = blockIdx.y;
  const int t0 = threadIdx.x << 3;
  const __bf16* xp = xbf + ((size_t)b << 18) + ((size_t)(half << 8) << 9) + t0;
  const float* ap = aw + ((size_t)b << 9) + (half << 8);
  float a0 = 0, a1 = 0, a2 = 0, a3 = 0, a4 = 0, a5 = 0, a6 = 0, a7 = 0;
  for (int m = 0; m < 256; ++m) {
    float w = ap[m];                               // uniform -> scalar load
    u32x4 v = *(const u32x4*)(xp + ((size_t)m << 9));
    a0 = fmaf(w, lo2f(v[0]), a0); a1 = fmaf(w, hi2f(v[0]), a1);
    a2 = fmaf(w, lo2f(v[1]), a2); a3 = fmaf(w, hi2f(v[1]), a3);
    a4 = fmaf(w, lo2f(v[2]), a4); a5 = fmaf(w, hi2f(v[2]), a5);
    a6 = fmaf(w, lo2f(v[3]), a6); a7 = fmaf(w, hi2f(v[3]), a7);
  }
  float* o = ctxp + (size_t)half * (NB * NT) + ((size_t)b << 9) + t0;
  o[0] = a0; o[1] = a1; o[2] = a2; o[3] = a3;
  o[4] = a4; o[5] = a5; o[6] = a6; o[7] = a7;
}

// ------------- ctx = sum of halves -> bf16 into acomb[:, 512:] -------------
__global__ __launch_bounds__(256) void ctx_combine(const float* __restrict__ ctxp,
                                                   __bf16* __restrict__ acomb) {
  int idx = blockIdx.x * 256 + threadIdx.x;       // 131072
  int b = idx >> 9, tp = idx & 511;
  float c = ctxp[idx] + ctxp[NB * NT + idx];
  acomb[((size_t)b << 10) + 512 + tp] = f2bf(c);
}

// ----------------------------- GRU pointwise update ------------------------
__global__ __launch_bounds__(256) void gru_update(const float* __restrict__ gi,
                                                  const float* __restrict__ gh,
                                                  float* __restrict__ h,
                                                  __bf16* __restrict__ axh) {
  int idx = blockIdx.x * 256 + threadIdx.x;       // 131072
  int b = idx >> 9, j = idx & 511;
  const float* gib = gi + (size_t)b * 1536;
  const float* ghb = gh + (size_t)b * 1536;
  float r = sigf(gib[j] + ghb[j]);
  float z = sigf(gib[j + 512] + ghb[j + 512]);
  float n = tanhf(gib[j + 1024] + r * ghb[j + 1024]);
  float hn = (1.0f - z) * n + z * h[idx];
  h[idx] = hn;
  axh[((size_t)b << 10) + 512 + j] = f2bf(hn);    // bf16 h for next-step GEMMs
}

// ===========================================================================
extern "C" void kernel_launch(void* const* d_in, const int* in_sizes, int n_in,
                              void* d_out, int out_size, void* d_ws, size_t ws_size,
                              hipStream_t stream) {
  const float* x      = (const float*)d_in[0];
  const float* W_attn = (const float*)d_in[1];
  const float* b_attn = (const float*)d_in[2];
  const float* W_comb = (const float*)d_in[3];
  const float* b_comb = (const float*)d_in[4];
  const float* W_ih   = (const float*)d_in[5];
  const float* W_hh   = (const float*)d_in[6];
  const float* b_ih   = (const float*)d_in[7];
  const float* b_hh   = (const float*)d_in[8];
  const float* W_out  = (const float*)d_in[9];
  const float* b_out  = (const float*)d_in[10];

  // ---- workspace carve (all chunks 256B-aligned multiples) ----
  char* p = (char*)d_ws;
  __bf16* xbf   = (__bf16*)p; p += (size_t)NB * NH * NT * 2;   // 134,217,728
  __bf16* axh   = (__bf16*)p; p += (size_t)NB * 1024 * 2;      // [x_t | h] bf16
  __bf16* acomb = (__bf16*)p; p += (size_t)NB * 1024 * 2;      // [x_t | ctx] bf16
  float*  s     = (float*)p;  p += (size_t)NB * NT * 4;        // attn logits / aw
  float*  ctxp  = (float*)p;  p += (size_t)2 * NB * NT * 4;    // einsum partials
  __bf16* g     = (__bf16*)p; p += (size_t)NB * NH * 2;
  float*  gi    = (float*)p;  p += (size_t)NB * 1536 * 4;
  float*  gh    = (float*)p;  p += (size_t)NB * 1536 * 4;
  float*  h     = (float*)p;  p += (size_t)NB * NH * 4;
  __bf16* wattn = (__bf16*)p; p += (size_t)NT * 1024 * 2;
  __bf16* wcomb = (__bf16*)p; p += (size_t)NH * 1024 * 2;
  __bf16* wih   = (__bf16*)p; p += (size_t)1536 * NH * 2;
  __bf16* whh   = (__bf16*)p; p += (size_t)1536 * NH * 2;
  __bf16* wout  = (__bf16*)p; p += (size_t)NH * NH * 2;

  // ---- prologue: bf16 conversions + state init ----
  cvt4_f32_bf16<<<(NB * NH * NT / 4 + 255) / 256, 256, 0, stream>>>(x, xbf, NB * NH * NT / 4);
  cvt4_f32_bf16<<<512, 256, 0, stream>>>(W_attn, wattn, NT * 1024 / 4);
  cvt4_f32_bf16<<<512, 256, 0, stream>>>(W_comb, wcomb, NH * 1024 / 4);
  cvt4_f32_bf16<<<768, 256, 0, stream>>>(W_ih, wih, 1536 * NH / 4);
  cvt4_f32_bf16<<<768, 256, 0, stream>>>(W_hh, whh, 1536 * NH / 4);
  cvt4_f32_bf16<<<256, 256, 0, stream>>>(W_out, wout, NH * NH / 4);
  init_state<<<512, 256, 0, stream>>>(h, axh);

  // ---- sequential scan over T ----
  for (int t = 0; t < NT; ++t) {
    prep_xt<<<NB, 256, 0, stream>>>(xbf, axh, acomb, t);
    // attn logits: [256,512] = axh[256,1024] x W_attn[512,1024].T
    gemm_wmma<<<16, 256, 0, stream>>>(axh, 1024, wattn, b_attn, s, 512,
                                      NB, 512, 1024, 0);
    softmax512<<<NB, 256, 0, stream>>>(s);
    einsum_ctx<<<dim3(NB, 2), 64, 0, stream>>>(s, xbf, ctxp);
    ctx_combine<<<512, 256, 0, stream>>>(ctxp, acomb);
    // g = relu(acomb x W_comb.T + b_comb) -> bf16
    gemm_wmma<<<16, 256, 0, stream>>>(acomb, 1024, wcomb, b_comb, g, 512,
                                      NB, 512, 1024, 3);
    // gi = g x W_ih.T + b_ih ; gh = h x W_hh.T + b_hh   (f32 out)
    gemm_wmma<<<48, 256, 0, stream>>>(g, 512, wih, b_ih, gi, 1536,
                                      NB, 1536, 512, 0);
    gemm_wmma<<<48, 256, 0, stream>>>(axh + 512, 1024, whh, b_hh, gh, 1536,
                                      NB, 1536, 512, 0);
    gru_update<<<512, 256, 0, stream>>>(gi, gh, h, axh);
  }

  // ---- epilogue: out = hT x W_out.T + b_out (f32) ----
  gemm_wmma<<<16, 256, 0, stream>>>(axh + 512, 1024, wout, b_out, (float*)d_out, 512,
                                    NB, 512, 512, 0);
}
